// SparseCausalSelfAttention_1803886265463
// MI455X (gfx1250) — compile-verified, matrix operand order checked by reading, not run on hardware
//
#include <hip/hip_runtime.h>
#include <hip/hip_bf16.h>

typedef __bf16 bf16_t;
typedef __attribute__((ext_vector_type(16))) __bf16        bf16x16;
typedef __attribute__((ext_vector_type(8)))  float         f32x8;
typedef __attribute__((ext_vector_type(4)))  unsigned int  u32x4;

#define NB   2
#define NT   2048
#define NE   2048
#define NH   16
#define HSZ  128
#define C3   6144   // 3 * NE

// ---------------------------------------------------------------------------
// WMMA fragment helpers (CDNA5 16x16x32 bf16, wave32).
// A fragment: lane L (L<16): row = L, ks = {0..7, 16..23}; L>=16: row = L-16,
// ks = {8..15, 24..31}.  => two contiguous 16-byte loads at k and k+16.
// B fragment mirrors A with row->column (we store B operands N-major).
// C/D fragment: lane L: col = L&15, rows = (L>>4)*8 + {0..7}.
// ---------------------------------------------------------------------------
static __device__ __forceinline__ bf16x16 frag_ld_row(const bf16_t* rowp, int k, int lane) {
  const bf16_t* p = rowp + k + ((lane >> 4) << 3);
  union { u32x4 u[2]; bf16x16 v; } f;
  f.u[0] = *reinterpret_cast<const u32x4*>(p);
  f.u[1] = *reinterpret_cast<const u32x4*>(p + 16);
  return f.v;
}
static __device__ __forceinline__ bf16x16 frag_ld(const bf16_t* tile, int ld, int k, int lane) {
  return frag_ld_row(tile + (lane & 15) * ld, k, lane);
}
static __device__ __forceinline__ f32x8 wmma_bf16(bf16x16 a, bf16x16 b, f32x8 c) {
  return __builtin_amdgcn_wmma_f32_16x16x32_bf16(false, a, false, b, (short)0, c, false, false);
}

// CDNA5 async DMA: global -> LDS, no VGPR data, tracked by ASYNCcnt.
// VDST carries the LDS byte address.
static __device__ __forceinline__ void async_ld_b128(unsigned lds_addr, const void* gsrc) {
  asm volatile("global_load_async_to_lds_b128 %0, %1, off"
               :: "v"(lds_addr), "v"(gsrc) : "memory");
}
static __device__ __forceinline__ void wait_async0() {
  asm volatile("s_wait_asynccnt 0" ::: "memory");
}
static __device__ __forceinline__ unsigned lds_addr_of(const void* p) {
  return (unsigned)(size_t)p;   // low 32 bits of generic LDS ptr = LDS offset
}

// ---------------------------------------------------------------------------
// fp32 -> bf16 conversion (vectorized x4)
// ---------------------------------------------------------------------------
__global__ __launch_bounds__(256)
void cvt_bf16_kernel(const float* __restrict__ in, bf16_t* __restrict__ out, int n4) {
  int i = blockIdx.x * 256 + threadIdx.x;
  if (i < n4) {
    float4 v = reinterpret_cast<const float4*>(in)[i];
    union { bf16_t b[4]; unsigned long long u; } p;
    p.b[0] = (bf16_t)v.x; p.b[1] = (bf16_t)v.y;
    p.b[2] = (bf16_t)v.z; p.b[3] = (bf16_t)v.w;
    reinterpret_cast<unsigned long long*>(out)[i] = p.u;
  }
}

// fp32 [R][C] -> bf16 [C][R]  (N-major weights for B fragments)
__global__ __launch_bounds__(256)
void transpose_bf16_kernel(const float* __restrict__ in, bf16_t* __restrict__ out, int R, int C) {
  __shared__ float tile[32][33];
  int c0 = blockIdx.x * 32, r0 = blockIdx.y * 32;
  int tx = threadIdx.x & 31, ty = threadIdx.x >> 5;  // 32 x 8
  #pragma unroll
  for (int j = 0; j < 32; j += 8)
    tile[ty + j][tx] = in[(size_t)(r0 + ty + j) * C + c0 + tx];
  __syncthreads();
  #pragma unroll
  for (int j = 0; j < 32; j += 8)
    out[(size_t)(c0 + ty + j) * R + r0 + tx] = (bf16_t)tile[tx][ty + j];
}

// ---------------------------------------------------------------------------
// QKV GEMM + scatter.  X [2048][2048] bf16 (stream tokens), WT [6144][2048]
// bf16 (N-major).  Wave tile 32x64 (2 A frags x 4 B frags -> 8 WMMA/k-step).
// Writes Q,K head-major [B,H,T,HS]; V transposed [B,H,HS,T].
// Stream row r -> batch r>>10, token t = 2*(r&1023) + sbit.
// ---------------------------------------------------------------------------
__global__ __launch_bounds__(128)
void qkv_gemm_kernel(const bf16_t* __restrict__ X, const bf16_t* __restrict__ WT,
                     bf16_t* __restrict__ Q, bf16_t* __restrict__ K,
                     bf16_t* __restrict__ Vt, int sbit) {
  const int lane = threadIdx.x & 31;
  const int wave = threadIdx.x >> 5;
  const int m0 = blockIdx.x * 32;
  const int n0 = blockIdx.y * 256 + wave * 64;
  f32x8 acc[2][4] = {};
  for (int k = 0; k < NE; k += 32) {
    bf16x16 a0 = frag_ld(X + (size_t)m0 * NE,        NE, k, lane);
    bf16x16 a1 = frag_ld(X + (size_t)(m0 + 16) * NE, NE, k, lane);
    #pragma unroll
    for (int j = 0; j < 4; ++j) {
      bf16x16 b = frag_ld(WT + (size_t)(n0 + j * 16) * NE, NE, k, lane);
      acc[0][j] = wmma_bf16(a0, b, acc[0][j]);
      acc[1][j] = wmma_bf16(a1, b, acc[1][j]);
    }
  }
  const int nl = lane & 15, rb = (lane >> 4) * 8;
  #pragma unroll
  for (int mi = 0; mi < 2; ++mi) {
    #pragma unroll
    for (int j = 0; j < 4; ++j) {
      int col = n0 + j * 16 + nl;
      int g = col >> 11;          // 0=q,1=k,2=v
      int e = col & 2047;
      int h = e >> 7, d = e & 127;
      #pragma unroll
      for (int i = 0; i < 8; ++i) {
        int r = m0 + mi * 16 + rb + i;
        int bb = r >> 10;
        int t  = ((r & 1023) << 1) | sbit;
        bf16_t val = (bf16_t)acc[mi][j][i];
        if (g == 0)      Q [((size_t)(bb * NH + h) * NT + t) * HSZ + d] = val;
        else if (g == 1) K [((size_t)(bb * NH + h) * NT + t) * HSZ + d] = val;
        else             Vt[((size_t)(bb * NH + h) * HSZ + d) * NT + t] = val;
      }
    }
  }
}

// ---------------------------------------------------------------------------
// RoPE in-place on Q and K (fp32 math). One thread per (b,h,t,d<64).
// ---------------------------------------------------------------------------
__global__ __launch_bounds__(256)
void rope_kernel(bf16_t* __restrict__ Q, bf16_t* __restrict__ K,
                 const float* __restrict__ cosb, const float* __restrict__ sinb) {
  int idx = blockIdx.x * 256 + threadIdx.x;       // NB*NH*NT*64 threads
  int d  = idx & 63;
  int t  = (idx >> 6) & (NT - 1);
  int bh = idx >> 17;
  bf16_t* q = Q + ((size_t)bh * NT + t) * HSZ;
  bf16_t* k = K + ((size_t)bh * NT + t) * HSZ;
  float c = cosb[t * HSZ + d], s = sinb[t * HSZ + d];
  float q1 = (float)q[d], q2 = (float)q[d + 64];
  q[d]      = (bf16_t)(q1 * c - q2 * s);
  q[d + 64] = (bf16_t)(q2 * c + q1 * s);
  float k1 = (float)k[d], k2 = (float)k[d + 64];
  k[d]      = (bf16_t)(k1 * c - k2 * s);
  k[d + 64] = (bf16_t)(k2 * c + k1 * s);
}

// ---------------------------------------------------------------------------
// Flash attention with double-buffered async-DMA K/V tiles.
// Block = 128 threads (4 waves); wave owns 16 query rows.
// grid = (T/64, H, B). Q,K [B,H,T,HS]; Vt [B,H,HS,T]; Y [B*T][NE] bf16.
// ---------------------------------------------------------------------------
__global__ __launch_bounds__(128)
void flash_attn_kernel(const bf16_t* __restrict__ Q, const bf16_t* __restrict__ K,
                       const bf16_t* __restrict__ Vt, bf16_t* __restrict__ Y) {
  __shared__ __align__(16) bf16_t Ks[2][64 * 128];   // [buf][key][d]
  __shared__ __align__(16) bf16_t Vs[2][128 * 64];   // [buf][d][key] (pre-transposed V)
  __shared__ __align__(16) bf16_t Ps[4][16 * 64];    // per-wave P staging

  const int lane = threadIdx.x & 31;
  const int wave = threadIdx.x >> 5;
  const int qt = blockIdx.x, h = blockIdx.y, b = blockIdx.z;
  const float scale = 0.08838834764831845f;          // 1/sqrt(128)

  const bf16_t* Qb = Q  + (size_t)(b * NH + h) * NT * HSZ;
  const bf16_t* Kb = K  + (size_t)(b * NH + h) * NT * HSZ;
  const bf16_t* Vb = Vt + (size_t)(b * NH + h) * HSZ * NT;

  // per-thread DMA slice: 128B of K tile + 128B of V tile
  const int krow  = threadIdx.x >> 1;
  const int khalf = (threadIdx.x & 1) * 64;
  const unsigned kdst0 = lds_addr_of(&Ks[0][0]) + (unsigned)(krow * 128 + khalf) * 2u;
  const unsigned vdst0 = lds_addr_of(&Vs[0][0]) + (unsigned)(threadIdx.x * 64) * 2u;
  const unsigned bufstride = (unsigned)(64 * 128 * 2);   // == sizeof one K (and V) buffer

  auto issue_tile = [&](int kt, int buf) {
    const bf16_t* ksrc = Kb + (size_t)(kt * 64 + krow) * HSZ + khalf;
    const bf16_t* vsrc = Vb + (size_t)threadIdx.x * NT + kt * 64;
    unsigned kd = kdst0 + buf * bufstride;
    unsigned vd = vdst0 + buf * bufstride;
    #pragma unroll
    for (int j = 0; j < 8; ++j) {
      async_ld_b128(kd + j * 16, ksrc + j * 8);
      async_ld_b128(vd + j * 16, vsrc + j * 8);
    }
  };

  const int qrow = qt * 64 + wave * 16;
  const bf16_t* qrp = Qb + (size_t)(qrow + (lane & 15)) * HSZ;
  bf16x16 aq[4];
  #pragma unroll
  for (int j = 0; j < 4; ++j) aq[j] = frag_ld_row(qrp, j * 32, lane);

  f32x8 o[8] = {};
  float m_i[8], l_i[8];
  #pragma unroll
  for (int i = 0; i < 8; ++i) { m_i[i] = -3.0e30f; l_i[i] = 0.0f; }

  issue_tile(0, 0);
  wait_async0();
  __syncthreads();

  for (int kt = 0; kt <= qt; ++kt) {
    const int cur = kt & 1;
    if (kt + 1 <= qt) issue_tile(kt + 1, cur ^ 1);  // DMA next tile during compute

    const bf16_t* KsT = &Ks[cur][0];
    const bf16_t* VsT = &Vs[cur][0];

    // S = Q K^T  (4 key subtiles of 16)
    f32x8 s[4];
    #pragma unroll
    for (int st = 0; st < 4; ++st) {
      f32x8 acc = {};
      #pragma unroll
      for (int j = 0; j < 4; ++j) {
        bf16x16 bk = frag_ld(KsT + st * 16 * 128, 128, j * 32, lane);
        acc = wmma_bf16(aq[j], bk, acc);
      }
      s[st] = acc;
    }

    // online softmax (rows split across 16-lane halves -> xor-shuffle reduce)
    const int kbase = kt * 64;
    float mtile[8];
    #pragma unroll
    for (int i = 0; i < 8; ++i) {
      int tq = qrow + (lane >> 4) * 8 + i;
      float mx = -3.0e30f;
      #pragma unroll
      for (int st = 0; st < 4; ++st) {
        int kg = kbase + st * 16 + (lane & 15);
        float v = s[st][i] * scale;
        v = (kg <= tq) ? v : -3.0e30f;
        s[st][i] = v;
        mx = fmaxf(mx, v);
      }
      mx = fmaxf(mx, __shfl_xor(mx, 1));
      mx = fmaxf(mx, __shfl_xor(mx, 2));
      mx = fmaxf(mx, __shfl_xor(mx, 4));
      mx = fmaxf(mx, __shfl_xor(mx, 8));
      mtile[i] = mx;
    }

    bf16_t* pw = &Ps[wave][0];
    #pragma unroll
    for (int i = 0; i < 8; ++i) {
      float mnew  = fmaxf(m_i[i], mtile[i]);
      float alpha = __expf(m_i[i] - mnew);
      m_i[i] = mnew;
      float ssum = 0.0f;
      #pragma unroll
      for (int st = 0; st < 4; ++st) {
        float p = __expf(s[st][i] - mnew);
        s[st][i] = p;
        ssum += p;
        pw[((lane >> 4) * 8 + i) * 64 + st * 16 + (lane & 15)] = (bf16_t)p;
      }
      ssum += __shfl_xor(ssum, 1);
      ssum += __shfl_xor(ssum, 2);
      ssum += __shfl_xor(ssum, 4);
      ssum += __shfl_xor(ssum, 8);
      l_i[i] = l_i[i] * alpha + ssum;
      #pragma unroll
      for (int j = 0; j < 8; ++j) o[j][i] *= alpha;
    }

    // O += P V
    bf16x16 ap0 = frag_ld(pw, 64, 0,  lane);
    bf16x16 ap1 = frag_ld(pw, 64, 32, lane);
    #pragma unroll
    for (int nt = 0; nt < 8; ++nt) {
      bf16x16 bv0 = frag_ld(VsT + nt * 16 * 64, 64, 0,  lane);
      bf16x16 bv1 = frag_ld(VsT + nt * 16 * 64, 64, 32, lane);
      o[nt] = wmma_bf16(ap0, bv0, o[nt]);
      o[nt] = wmma_bf16(ap1, bv1, o[nt]);
    }

    wait_async0();       // next tile's DMA has landed
    __syncthreads();     // everyone done with cur buffer before it is reused
  }

  // epilogue: O /= l, write Y[b*T+t][h*128+d]
  #pragma unroll
  for (int i = 0; i < 8; ++i) {
    float inv = 1.0f / l_i[i];
    int t = qrow + (lane >> 4) * 8 + i;
    bf16_t* yr = Y + ((size_t)(b * NT + t)) * NE + h * HSZ;
    #pragma unroll
    for (int nt = 0; nt < 8; ++nt)
      yr[nt * 16 + (lane & 15)] = (bf16_t)(o[nt][i] * inv);
  }
}

// ---------------------------------------------------------------------------
// Output projection per stream: gather interleaved token rows from Y,
// wave tile 32x64, GEMM against N-major WprojT, fp32 out.
// ---------------------------------------------------------------------------
__global__ __launch_bounds__(128)
void proj_gemm_kernel(const bf16_t* __restrict__ Y, const bf16_t* __restrict__ WT,
                      float* __restrict__ out, int sbit) {
  const int lane = threadIdx.x & 31;
  const int wave = threadIdx.x >> 5;
  const int m0 = blockIdx.x * 32;
  const int n0 = blockIdx.y * 256 + wave * 64;
  int r0 = m0 + (lane & 15);
  int r1 = r0 + 16;
  const bf16_t* arow0 =
      Y + (size_t)((r0 >> 10) * NT + (((r0 & 1023) << 1) | sbit)) * NE;
  const bf16_t* arow1 =
      Y + (size_t)((r1 >> 10) * NT + (((r1 & 1023) << 1) | sbit)) * NE;
  f32x8 acc[2][4] = {};
  for (int k = 0; k < NE; k += 32) {
    bf16x16 a0 = frag_ld_row(arow0, k, lane);
    bf16x16 a1 = frag_ld_row(arow1, k, lane);
    #pragma unroll
    for (int j = 0; j < 4; ++j) {
      bf16x16 b = frag_ld(WT + (size_t)(n0 + j * 16) * NE, NE, k, lane);
      acc[0][j] = wmma_bf16(a0, b, acc[0][j]);
      acc[1][j] = wmma_bf16(a1, b, acc[1][j]);
    }
  }
  const int nl = lane & 15, rb = (lane >> 4) * 8;
  #pragma unroll
  for (int mi = 0; mi < 2; ++mi)
    #pragma unroll
    for (int j = 0; j < 4; ++j)
      #pragma unroll
      for (int i = 0; i < 8; ++i)
        out[(size_t)(m0 + mi * 16 + rb + i) * NE + n0 + j * 16 + nl] = acc[mi][j][i];
}

// ---------------------------------------------------------------------------
extern "C" void kernel_launch(void* const* d_in, const int* in_sizes, int n_in,
                              void* d_out, int out_size, void* d_ws, size_t ws_size,
                              hipStream_t stream) {
  const float* x_a     = (const float*)d_in[0];
  const float* x_b     = (const float*)d_in[1];
  const float* cosb    = (const float*)d_in[2];
  const float* sinb    = (const float*)d_in[3];
  // d_in[4] = mask_a (deterministic even/odd -> not needed)
  const float* Wqkv_a  = (const float*)d_in[5];
  const float* Wqkv_b  = (const float*)d_in[6];
  const float* Wproj_a = (const float*)d_in[7];
  const float* Wproj_b = (const float*)d_in[8];
  float* out = (float*)d_out;

  // workspace bump allocator
  char* ws = (char*)d_ws;
  size_t off = 0;
  auto alloc = [&](size_t bytes) { char* p = ws + off; off = (off + bytes + 255) & ~(size_t)255; return p; };
  bf16_t* Xa      = (bf16_t*)alloc((size_t)2048 * NE * 2);
  bf16_t* Xb      = (bf16_t*)alloc((size_t)2048 * NE * 2);
  bf16_t* WqkvaT  = (bf16_t*)alloc((size_t)C3 * NE * 2);
  bf16_t* WqkvbT  = (bf16_t*)alloc((size_t)C3 * NE * 2);
  bf16_t* WprojaT = (bf16_t*)alloc((size_t)NE * NE * 2);
  bf16_t* WprojbT = (bf16_t*)alloc((size_t)NE * NE * 2);
  bf16_t* Qbuf    = (bf16_t*)alloc((size_t)NB * NH * NT * HSZ * 2);
  bf16_t* Kbuf    = (bf16_t*)alloc((size_t)NB * NH * NT * HSZ * 2);
  bf16_t* Vtbuf   = (bf16_t*)alloc((size_t)NB * NH * HSZ * NT * 2);
  bf16_t* Ybuf    = (bf16_t*)alloc((size_t)NB * NT * NE * 2);

  // 1) convert activations to bf16
  {
    int n4 = (2048 * NE) / 4;
    cvt_bf16_kernel<<<(n4 + 255) / 256, 256, 0, stream>>>(x_a, Xa, n4);
    cvt_bf16_kernel<<<(n4 + 255) / 256, 256, 0, stream>>>(x_b, Xb, n4);
  }
  // 2) transpose-convert weights to N-major bf16
  transpose_bf16_kernel<<<dim3(C3 / 32, NE / 32), 256, 0, stream>>>(Wqkv_a, WqkvaT, NE, C3);
  transpose_bf16_kernel<<<dim3(C3 / 32, NE / 32), 256, 0, stream>>>(Wqkv_b, WqkvbT, NE, C3);
  transpose_bf16_kernel<<<dim3(NE / 32, NE / 32), 256, 0, stream>>>(Wproj_a, WprojaT, NE, NE);
  transpose_bf16_kernel<<<dim3(NE / 32, NE / 32), 256, 0, stream>>>(Wproj_b, WprojbT, NE, NE);
  // 3) QKV GEMMs with head-major scatter (V pre-transposed)
  qkv_gemm_kernel<<<dim3(2048 / 32, C3 / 256), 128, 0, stream>>>(Xa, WqkvaT, Qbuf, Kbuf, Vtbuf, 0);
  qkv_gemm_kernel<<<dim3(2048 / 32, C3 / 256), 128, 0, stream>>>(Xb, WqkvbT, Qbuf, Kbuf, Vtbuf, 1);
  // 4) RoPE on Q and K
  {
    int n = NB * NH * NT * 64;
    rope_kernel<<<n / 256, 256, 0, stream>>>(Qbuf, Kbuf, cosb, sinb);
  }
  // 5) flash attention (async-DMA double-buffered tiles)
  flash_attn_kernel<<<dim3(NT / 64, NH, NB), 128, 0, stream>>>(Qbuf, Kbuf, Vtbuf, Ybuf);
  // 6) output projections (stream a then b, concatenated in d_out)
  proj_gemm_kernel<<<dim3(2048 / 32, NE / 256), 128, 0, stream>>>(Ybuf, WprojaT, out, 0);
  proj_gemm_kernel<<<dim3(2048 / 32, NE / 256), 128, 0, stream>>>(Ybuf, WprojbT, out + (size_t)2048 * NE, 1);
}
